// ToolSelectNet_50165218017747
// MI455X (gfx1250) — matrix operand order repaired and checked by reading, not compiled, not run on hardware
//
#include <hip/hip_runtime.h>
#include <hip/hip_bf16.h>
#include <math.h>

typedef __attribute__((ext_vector_type(16))) _Float16 v16h;
typedef __attribute__((ext_vector_type(8)))  _Float16 v8h_t;
typedef __attribute__((ext_vector_type(8)))  float    v8f;

// ---------------------------------------------------------------------------
// CDNA5 async global->LDS helpers (inline asm; portable across toolchains).
// global_load_async_to_lds_b128: VDST = LDS byte offset VGPR, VADDR = 64-bit
// global address. Tracked by ASYNCcnt; drained with s_wait_asynccnt.
// ---------------------------------------------------------------------------
__device__ __forceinline__ void async_load_lds_b128(unsigned lds_off, const void* gaddr) {
  asm volatile("global_load_async_to_lds_b128 %0, %1, off"
               :: "v"(lds_off), "v"((unsigned long long)(uintptr_t)gaddr)
               : "memory");
}
__device__ __forceinline__ void wait_asynccnt0() {
  asm volatile("s_wait_asynccnt 0x0" ::: "memory");
}

// ---------------------------------------------------------------------------
// Pass 1: single read of the 205MB input.
//   rgb = [c0, 0.5*(c1+c2), c3]; area-resize 448->112 (4x4 mean) -> f16
//   gray = 0.299 r + 0.587 g + 0.114 b at full 448x448 -> f16
// ---------------------------------------------------------------------------
__global__ __launch_bounds__(256) void preprocess_kernel(const float* __restrict__ img,
                                                         _Float16* __restrict__ rgbs,
                                                         _Float16* __restrict__ gray) {
  const int P = 112 * 112;
  int t = blockIdx.x * 256 + threadIdx.x;
  if (t >= 64 * P) return;
  int b = t / P, p = t % P;
  int oy = p / 112, ox = p % 112;
  const float* ib = img + (size_t)b * 4 * 200704;
  _Float16* gb = gray + (size_t)b * 200704;
  float rs = 0.f, gs = 0.f, bs = 0.f;
#pragma unroll
  for (int dy = 0; dy < 4; ++dy) {
    int iy = oy * 4 + dy;
#pragma unroll
    for (int dx = 0; dx < 4; ++dx) {
      int ix = ox * 4 + dx;
      int off = iy * 448 + ix;
      float c0 = ib[off];
      float c1 = ib[200704 + off];
      float c2 = ib[2 * 200704 + off];
      float c3 = ib[3 * 200704 + off];
      float g2 = 0.5f * (c1 + c2);
      rs += c0; gs += g2; bs += c3;
      gb[off] = (_Float16)(0.299f * c0 + 0.587f * g2 + 0.114f * c3);
    }
  }
  rgbs[((size_t)b * 3 + 0) * P + p] = (_Float16)(rs * (1.f / 16.f));
  rgbs[((size_t)b * 3 + 1) * P + p] = (_Float16)(gs * (1.f / 16.f));
  rgbs[((size_t)b * 3 + 2) * P + p] = (_Float16)(bs * (1.f / 16.f));
}

__global__ __launch_bounds__(256) void zero_u32_kernel(unsigned* __restrict__ p, int n) {
  int t = blockIdx.x * 256 + threadIdx.x;
  if (t < n) p[t] = 0u;
}

// ---------------------------------------------------------------------------
// Intensity histograms: one block per image, integer LDS atomics (ds_add).
// ---------------------------------------------------------------------------
__global__ __launch_bounds__(256) void inten_hist_kernel(const _Float16* __restrict__ gray,
                                                         float* __restrict__ hist) {
  int img = blockIdx.x, tid = threadIdx.x;
  __shared__ unsigned hs[4][32];
  if (tid < 128) ((unsigned*)hs)[tid] = 0u;
  __syncthreads();
  const _Float16* gb = gray + (size_t)img * 200704;
  for (int t = tid; t < 200704; t += 256) {
    int y = t / 448, x = t % 448;
    float g = (float)gb[t];
    int bin = (int)(g * 32.f);
    bin = bin < 0 ? 0 : (bin > 31 ? 31 : bin);
    int q = (y / 224) * 2 + (x / 224);
    atomicAdd(&hs[q][bin], 1u);
  }
  __syncthreads();
  if (tid < 32) {
    unsigned tot = hs[0][tid] + hs[1][tid] + hs[2][tid] + hs[3][tid];
    float* hb = hist + img * 160;
    hb[tid] = (float)tot * (1.f / 200704.f);
#pragma unroll
    for (int q = 0; q < 4; ++q)
      hb[(1 + q) * 32 + tid] = (float)hs[q][tid] * (1.f / 50176.f);
  }
}

// ---------------------------------------------------------------------------
// Sobel magnitude (kornia /8 normalization, replicate pad) + histogram.
// 32x32 tiles with 34x34 LDS halo; each tile sits in exactly one quadrant.
// ---------------------------------------------------------------------------
__global__ __launch_bounds__(256) void sobel_hist_kernel(const _Float16* __restrict__ gray,
                                                         unsigned* __restrict__ counts) {
  int img = blockIdx.y, tile = blockIdx.x, tid = threadIdx.x;
  int ty0 = (tile / 14) * 32, tx0 = (tile % 14) * 32;
  __shared__ float s[34][34];
  __shared__ unsigned hs[32];
  if (tid < 32) hs[tid] = 0u;
  const _Float16* gb = gray + (size_t)img * 200704;
  for (int l = tid; l < 34 * 34; l += 256) {
    int ly = l / 34, lx = l % 34;
    int y = ty0 - 1 + ly; y = y < 0 ? 0 : (y > 447 ? 447 : y);
    int x = tx0 - 1 + lx; x = x < 0 ? 0 : (x > 447 ? 447 : x);
    s[ly][lx] = (float)gb[y * 448 + x];
  }
  __syncthreads();
  int px = tid & 31, y0 = tid >> 5;
#pragma unroll
  for (int r = 0; r < 4; ++r) {
    int ly = y0 + r * 8 + 1, lx = px + 1;
    float a00 = s[ly - 1][lx - 1], a01 = s[ly - 1][lx], a02 = s[ly - 1][lx + 1];
    float a10 = s[ly][lx - 1],                           a12 = s[ly][lx + 1];
    float a20 = s[ly + 1][lx - 1], a21 = s[ly + 1][lx], a22 = s[ly + 1][lx + 1];
    float gx = (a02 - a00 + 2.f * (a12 - a10) + a22 - a20) * 0.125f;
    float gy = (a20 + 2.f * a21 + a22 - a00 - 2.f * a01 - a02) * 0.125f;
    float mag = sqrtf(gx * gx + gy * gy + 1e-6f);
    int bin = (int)(mag * 32.f);
    bin = bin < 0 ? 0 : (bin > 31 ? 31 : bin);
    atomicAdd(&hs[bin], 1u);
  }
  __syncthreads();
  if (tid < 32) {
    int q = (ty0 / 224) * 2 + (tx0 / 224);
    if (hs[tid]) atomicAdd(&counts[(img * 4 + q) * 32 + tid], hs[tid]);
  }
}

__global__ __launch_bounds__(32) void grad_finalize_kernel(const unsigned* __restrict__ counts,
                                                           float* __restrict__ hist) {
  int img = blockIdx.x, bin = threadIdx.x;
  unsigned c0 = counts[(img * 4 + 0) * 32 + bin];
  unsigned c1 = counts[(img * 4 + 1) * 32 + bin];
  unsigned c2 = counts[(img * 4 + 2) * 32 + bin];
  unsigned c3 = counts[(img * 4 + 3) * 32 + bin];
  float* hb = hist + img * 160;
  hb[bin] = (float)(c0 + c1 + c2 + c3) * (1.f / 200704.f);
  hb[32 + bin]  = (float)c0 * (1.f / 50176.f);
  hb[64 + bin]  = (float)c1 * (1.f / 50176.f);
  hb[96 + bin]  = (float)c2 * (1.f / 50176.f);
  hb[128 + bin] = (float)c3 * (1.f / 50176.f);
}

// ---------------------------------------------------------------------------
// f32 [rows][ktrue] weights -> f16 [rows][kpad] (zero K-padding for WMMA)
// ---------------------------------------------------------------------------
__global__ __launch_bounds__(256) void convert_pad_kernel(const float* __restrict__ src,
                                                          _Float16* __restrict__ dst,
                                                          int rows, int ktrue, int kpad) {
  int t = blockIdx.x * 256 + threadIdx.x;
  if (t >= rows * kpad) return;
  int r = t / kpad, k = t % kpad;
  dst[t] = (_Float16)(k < ktrue ? src[(size_t)r * ktrue + k] : 0.f);
}

// ---------------------------------------------------------------------------
// Implicit-GEMM conv via V_WMMA_F32_16X16X32_F16, M-tiled 4x per wave:
// one gathered B fragment (im2col) feeds 4 WMMAs (64 output channels).
// A-frag halves i -> K=(i/8)*16+(lane/16)*8+(i%8) : two b128 loads/row.
// B-frag halves i -> K=(i/8)*16+(lane/16)*8+(i%8), N=lane%16.
// D: M = r + 8*(lane/16), N = lane%16. Bias + ReLU fused on store.
// ---------------------------------------------------------------------------
template <int CIN, int KH, int KW, int STR, int PAD, int H, int W, int OH, int OW,
          int KPAD, int KTRUE>
__global__ __launch_bounds__(32) void conv_wmma_kernel(const _Float16* __restrict__ act,
                                                       const _Float16* __restrict__ wgt,
                                                       const float* __restrict__ bias,
                                                       _Float16* __restrict__ out, int Cout) {
  const int lane = threadIdx.x;
  const int g = lane >> 4, lm = lane & 15;
  const int mbase = blockIdx.x * 64;
  const int n = blockIdx.y * 16 + lm;
  const int OHW = OH * OW, HW = H * W, KHW = KH * KW;
  const int b = n / OHW, pos = n % OHW;
  const int oy = pos / OW, ox = pos % OW;
  const int iy0 = oy * STR - PAD, ix0 = ox * STR - PAD;
  const _Float16* actb = act + (size_t)b * CIN * HW;
  const _Float16* w0 = wgt + (size_t)(mbase + lm) * KPAD;
  v8f acc[4] = {{}, {}, {}, {}};
  for (int k0 = 0; k0 < KPAD; k0 += 32) {
    __builtin_prefetch(w0 + k0 + 64, 0, 0);  // global_prefetch_b8 on weight stream
    // im2col gather of B (shared across the 4 M-tiles)
    v16h bf;
#pragma unroll
    for (int i = 0; i < 16; ++i) {
      int k = k0 + (i >> 3) * 16 + g * 8 + (i & 7);
      _Float16 v = (_Float16)0.f;
      if (KTRUE == KPAD || k < KTRUE) {
        int ci = k / KHW, rr = k % KHW;
        int iy = iy0 + rr / KW, ix = ix0 + rr % KW;
        if (iy >= 0 && iy < H && ix >= 0 && ix < W)
          v = actb[(size_t)ci * HW + iy * W + ix];
      }
      bf[i] = v;
    }
#pragma unroll
    for (int mt = 0; mt < 4; ++mt) {
      const _Float16* ap = w0 + (size_t)mt * 16 * KPAD + k0 + g * 8;
      v8h_t alo = *(const v8h_t*)ap;
      v8h_t ahi = *(const v8h_t*)(ap + 16);
      v16h a;
#pragma unroll
      for (int i = 0; i < 8; ++i) { a[i] = alo[i]; a[i + 8] = ahi[i]; }
      acc[mt] = __builtin_amdgcn_wmma_f32_16x16x32_f16(false, a, false, bf, (short)0,
                                                       acc[mt], false, false);
    }
  }
#pragma unroll
  for (int mt = 0; mt < 4; ++mt) {
#pragma unroll
    for (int r = 0; r < 8; ++r) {
      int m = mbase + mt * 16 + r + 8 * g;
      float vv = acc[mt][r] + bias[m];
      vv = vv > 0.f ? vv : 0.f;
      out[((size_t)b * Cout + m) * OHW + pos] = (_Float16)vv;
    }
  }
}

// ---------------------------------------------------------------------------
// FC1 as TN WMMA GEMM with CDNA5 async global->LDS double-buffered staging.
// Block = 8 waves; wave w owns N-tile (blockIdx*8+w). The 64x32 A slab
// (f_cat rows, shared by ALL N-tiles) is staged per k-step into LDS with
// global_load_async_to_lds_b128 (ASYNCcnt), while B streams as direct b128
// loads. A fragments are then read back with ds_load_b128.
// ---------------------------------------------------------------------------
__global__ __launch_bounds__(256) void gemm_tn_async_kernel(const _Float16* __restrict__ A,
                                                            const _Float16* __restrict__ Bm,
                                                            const float* __restrict__ bias,
                                                            float* __restrict__ out,
                                                            int K, int N) {
  __shared__ __align__(16) _Float16 atile[2][64 * 32];  // two 4KB slabs
  const int tid = threadIdx.x, wave = tid >> 5, lane = tid & 31;
  const int g = lane >> 4, lm = lane & 15;
  const int n0 = (blockIdx.x * 8 + wave) * 16;
  const _Float16* brow = Bm + (size_t)(n0 + lm) * K;
  // staging map: thread t moves 16B; row = t/4 (M), chunk = t%4 (8 halves)
  const int srow = tid >> 2, schunk = tid & 3;
  const _Float16* gsrc = A + (size_t)srow * K + schunk * 8;
  const unsigned lds_st[2] = {
      (unsigned)(uintptr_t)&atile[0][srow * 32 + schunk * 8],
      (unsigned)(uintptr_t)&atile[1][srow * 32 + schunk * 8]};
  v8f acc[4] = {{}, {}, {}, {}};

  async_load_lds_b128(lds_st[0], gsrc);  // prologue: stage k0=0 into buf 0
  int buf = 0;
  for (int k0 = 0; k0 < K; k0 += 32) {
    wait_asynccnt0();        // this wave's staged slab has landed
    __syncthreads();         // everyone's slab portions have landed
    if (k0 + 32 < K)
      async_load_lds_b128(lds_st[buf ^ 1], gsrc + (k0 + 32));
    // B fragment: contiguous b128 pair from global
    const _Float16* bp = brow + k0 + g * 8;
    v8h_t blo = *(const v8h_t*)bp, bhi = *(const v8h_t*)(bp + 16);
    v16h bfrag;
#pragma unroll
    for (int i = 0; i < 8; ++i) { bfrag[i] = blo[i]; bfrag[i + 8] = bhi[i]; }
    const _Float16* at = &atile[buf][0];
#pragma unroll
    for (int mt = 0; mt < 4; ++mt) {
      const _Float16* ap = at + (mt * 16 + lm) * 32 + g * 8;  // ds_load_b128 x2
      v8h_t alo = *(const v8h_t*)ap;
      v8h_t ahi = *(const v8h_t*)(ap + 16);
      v16h a;
#pragma unroll
      for (int i = 0; i < 8; ++i) { a[i] = alo[i]; a[i + 8] = ahi[i]; }
      acc[mt] = __builtin_amdgcn_wmma_f32_16x16x32_f16(false, a, false, bfrag, (short)0,
                                                       acc[mt], false, false);
    }
    __syncthreads();         // all waves done reading buf before it is re-staged
    buf ^= 1;
  }
  int nn = n0 + lm;
#pragma unroll
  for (int mt = 0; mt < 4; ++mt) {
#pragma unroll
    for (int r = 0; r < 8; ++r) {
      int m = mt * 16 + r + 8 * g;
      float vv = acc[mt][r] + bias[nn];
      vv = vv > 0.f ? vv : 0.f;
      out[(size_t)m * N + nn] = vv;
    }
  }
}

// ---------------------------------------------------------------------------
// 3x3 stride-2 maxpool (VALID), f16 in, templated out type.
// ---------------------------------------------------------------------------
template <typename OutT>
__global__ __launch_bounds__(256) void maxpool_kernel(const _Float16* __restrict__ in,
                                                      OutT* __restrict__ out,
                                                      int BC, int H, int W, int OH, int OW) {
  int t = blockIdx.x * 256 + threadIdx.x;
  int tot = BC * OH * OW;
  if (t >= tot) return;
  int bc = t / (OH * OW), p = t % (OH * OW);
  int oy = p / OW, ox = p % OW;
  const _Float16* ib = in + (size_t)bc * H * W;
  float m = -1e30f;
#pragma unroll
  for (int dy = 0; dy < 3; ++dy)
#pragma unroll
    for (int dx = 0; dx < 3; ++dx) {
      float v = (float)ib[(oy * 2 + dy) * W + (ox * 2 + dx)];
      m = v > m ? v : m;
    }
  out[t] = (OutT)m;
}

// ---------------------------------------------------------------------------
// Feature assembly: per-image L2 norms + hist->[96][4][4] broadcast, f16 f_cat.
// ---------------------------------------------------------------------------
__device__ inline float histval(const float* h160, int ch, int s) {
  if (ch < 32) return h160[ch];
  if (ch < 64) {
    int i = s / 4, j = s % 4;
    int q = (i / 2) * 2 + (j / 2);
    return h160[(1 + q) * 32 + (ch - 32)];
  }
  return 0.f;
}

__global__ __launch_bounds__(256) void assemble_kernel(const float* __restrict__ p5,
                                                       const float* __restrict__ ihist,
                                                       const float* __restrict__ ghist,
                                                       _Float16* __restrict__ fcat) {
  int img = blockIdx.x, tid = threadIdx.x;
  const float* sem = p5 + (size_t)img * 1024;
  const float* ih = ihist + img * 160;
  const float* gh = ghist + img * 160;
  __shared__ float red[256];
  auto block_sum = [&](float x) -> float {
    red[tid] = x; __syncthreads();
    for (int s = 128; s > 0; s >>= 1) { if (tid < s) red[tid] += red[tid + s]; __syncthreads(); }
    float r = red[0]; __syncthreads(); return r;
  };
  float v = 0.f;
  for (int i = tid; i < 1024; i += 256) { float x = sem[i]; v += x * x; }
  float ss_sem = block_sum(v);
  v = 0.f;
  for (int t = tid; t < 160; t += 256) { float x = ih[t]; v += (t < 32 ? 16.f : 4.f) * x * x; }
  float ss_int = block_sum(v);
  v = 0.f;
  for (int t = tid; t < 160; t += 256) { float x = gh[t]; v += (t < 32 ? 16.f : 4.f) * x * x; }
  float ss_grad = block_sum(v);
  float inv_sem = 1.f / (sqrtf(ss_sem) + 1e-7f);
  float inv_int = 1.f / (sqrtf(ss_int) + 1e-7f);
  float inv_grad = 1.f / (sqrtf(ss_grad) + 1e-7f);
  _Float16* outv = fcat + (size_t)img * 4096;
  for (int i = tid; i < 1024; i += 256) outv[i] = (_Float16)(sem[i] * inv_sem);
  for (int j = tid; j < 1536; j += 256)
    outv[1024 + j] = (_Float16)(histval(ih, j / 16, j % 16) * inv_int);
  for (int j = tid; j < 1536; j += 256)
    outv[2560 + j] = (_Float16)(histval(gh, j / 16, j % 16) * inv_grad);
}

// ---------------------------------------------------------------------------
// FC2: 64x12 <- 64x256 @ 256x12^T + bias (tiny; scalar f32)
// ---------------------------------------------------------------------------
__global__ __launch_bounds__(32) void fc2_kernel(const float* __restrict__ h,
                                                 const float* __restrict__ w,
                                                 const float* __restrict__ bias,
                                                 float* __restrict__ out) {
  int b = blockIdx.x, j = threadIdx.x;
  if (j >= 12) return;
  const float* hb = h + b * 256;
  const float* wr = w + j * 256;
  float s = bias[j];
  for (int k = 0; k < 256; ++k) s += hb[k] * wr[k];
  out[b * 12 + j] = s;
}

// ---------------------------------------------------------------------------
extern "C" void kernel_launch(void* const* d_in, const int* in_sizes, int n_in,
                              void* d_out, int out_size, void* d_ws, size_t ws_size,
                              hipStream_t stream) {
  const float* image = (const float*)d_in[0];
  const float* w1 = (const float*)d_in[1];  const float* b1 = (const float*)d_in[2];
  const float* w2 = (const float*)d_in[3];  const float* b2 = (const float*)d_in[4];
  const float* w3 = (const float*)d_in[5];  const float* b3 = (const float*)d_in[6];
  const float* w4 = (const float*)d_in[7];  const float* b4 = (const float*)d_in[8];
  const float* w5 = (const float*)d_in[9];  const float* b5 = (const float*)d_in[10];
  const float* fc1w = (const float*)d_in[11]; const float* fc1b = (const float*)d_in[12];
  const float* fc2w = (const float*)d_in[13]; const float* fc2b = (const float*)d_in[14];
  float* out = (float*)d_out;

  char* wsp = (char*)d_ws;
  auto alloc = [&](size_t bytes) -> void* {
    void* p = (void*)wsp;
    wsp += (bytes + 255) & ~(size_t)255;
    return p;
  };
  _Float16* gray = (_Float16*)alloc(64ull * 200704 * 2);
  _Float16* rgbs = (_Float16*)alloc(64ull * 3 * 12544 * 2);
  _Float16* w1h = (_Float16*)alloc(64ull * 384 * 2);
  _Float16* w2h = (_Float16*)alloc(192ull * 1600 * 2);
  _Float16* w3h = (_Float16*)alloc(384ull * 1728 * 2);
  _Float16* w4h = (_Float16*)alloc(256ull * 3456 * 2);
  _Float16* w5h = (_Float16*)alloc(256ull * 2304 * 2);
  _Float16* fc1h = (_Float16*)alloc(256ull * 4096 * 2);
  _Float16* a1 = (_Float16*)alloc(64ull * 64 * 729 * 2);
  _Float16* p1 = (_Float16*)alloc(64ull * 64 * 169 * 2);
  _Float16* a2 = (_Float16*)alloc(64ull * 192 * 169 * 2);
  _Float16* p2 = (_Float16*)alloc(64ull * 192 * 36 * 2);
  _Float16* a3 = (_Float16*)alloc(64ull * 384 * 36 * 2);
  _Float16* a4 = (_Float16*)alloc(64ull * 256 * 36 * 2);
  _Float16* a5 = (_Float16*)alloc(64ull * 256 * 36 * 2);
  float* p5 = (float*)alloc(64ull * 1024 * 4);
  float* ihist = (float*)alloc(64ull * 160 * 4);
  float* ghist = (float*)alloc(64ull * 160 * 4);
  unsigned* gcounts = (unsigned*)alloc(64ull * 128 * 4);
  _Float16* fcat = (_Float16*)alloc(64ull * 4096 * 2);
  float* h1 = (float*)alloc(64ull * 256 * 4);

  // Stage 1: one pass over 205MB input
  preprocess_kernel<<<(64 * 12544 + 255) / 256, 256, 0, stream>>>(image, rgbs, gray);

  // Stage 2: histogram features (deterministic integer atomics)
  zero_u32_kernel<<<(64 * 128 + 255) / 256, 256, 0, stream>>>(gcounts, 64 * 128);
  inten_hist_kernel<<<64, 256, 0, stream>>>(gray, ihist);
  sobel_hist_kernel<<<dim3(196, 64), 256, 0, stream>>>(gray, gcounts);
  grad_finalize_kernel<<<64, 32, 0, stream>>>(gcounts, ghist);

  // Stage 3: weights -> f16, K zero-padded to multiples of 32
  convert_pad_kernel<<<(64 * 384 + 255) / 256, 256, 0, stream>>>(w1, w1h, 64, 363, 384);
  convert_pad_kernel<<<(192 * 1600 + 255) / 256, 256, 0, stream>>>(w2, w2h, 192, 1600, 1600);
  convert_pad_kernel<<<(384 * 1728 + 255) / 256, 256, 0, stream>>>(w3, w3h, 384, 1728, 1728);
  convert_pad_kernel<<<(256 * 3456 + 255) / 256, 256, 0, stream>>>(w4, w4h, 256, 3456, 3456);
  convert_pad_kernel<<<(256 * 2304 + 255) / 256, 256, 0, stream>>>(w5, w5h, 256, 2304, 2304);
  convert_pad_kernel<<<(256 * 4096 + 255) / 256, 256, 0, stream>>>(fc1w, fc1h, 256, 4096, 4096);

  // Stage 4: AlexNet conv stack as WMMA implicit GEMM (M-tiled 4x per wave)
  conv_wmma_kernel<3, 11, 11, 4, 2, 112, 112, 27, 27, 384, 363>
      <<<dim3(1, 2916), 32, 0, stream>>>(rgbs, w1h, b1, a1, 64);
  maxpool_kernel<_Float16><<<(64 * 64 * 169 + 255) / 256, 256, 0, stream>>>(a1, p1, 64 * 64, 27, 27, 13, 13);
  conv_wmma_kernel<64, 5, 5, 1, 2, 13, 13, 13, 13, 1600, 1600>
      <<<dim3(3, 676), 32, 0, stream>>>(p1, w2h, b2, a2, 192);
  maxpool_kernel<_Float16><<<(64 * 192 * 36 + 255) / 256, 256, 0, stream>>>(a2, p2, 64 * 192, 13, 13, 6, 6);
  conv_wmma_kernel<192, 3, 3, 1, 1, 6, 6, 6, 6, 1728, 1728>
      <<<dim3(6, 144), 32, 0, stream>>>(p2, w3h, b3, a3, 384);
  conv_wmma_kernel<384, 3, 3, 1, 1, 6, 6, 6, 6, 3456, 3456>
      <<<dim3(4, 144), 32, 0, stream>>>(a3, w4h, b4, a4, 256);
  conv_wmma_kernel<256, 3, 3, 1, 1, 6, 6, 6, 6, 2304, 2304>
      <<<dim3(4, 144), 32, 0, stream>>>(a4, w5h, b5, a5, 256);
  maxpool_kernel<float><<<(64 * 256 * 4 + 255) / 256, 256, 0, stream>>>(a5, p5, 64 * 256, 6, 6, 2, 2);

  // Stage 5: L2-normalized feature concat, FC1 (WMMA + async LDS), FC2
  assemble_kernel<<<64, 256, 0, stream>>>(p5, ihist, ghist, fcat);
  gemm_tn_async_kernel<<<dim3(2), 256, 0, stream>>>(fcat, fc1h, fc1b, h1, 4096, 256);
  fc2_kernel<<<64, 32, 0, stream>>>(h1, fc2w, fc2b, out);
}
